// SelfAttention_22170621182154
// MI455X (gfx1250) — matrix operand order repaired and checked by reading, not compiled, not run on hardware
//
#include <hip/hip_runtime.h>
#include <hip/hip_bf16.h>

#if defined(__has_builtin)
# if __has_builtin(__builtin_amdgcn_global_load_async_to_lds_b128) && \
     __has_builtin(__builtin_amdgcn_s_wait_asynccnt)
#  define HAS_ASYNC_LDS 1
# endif
#endif
#ifndef HAS_ASYNC_LDS
# define HAS_ASYNC_LDS 0
#endif

typedef __bf16 bf16;
typedef __attribute__((ext_vector_type(16))) __bf16 bf16x16;
typedef __attribute__((ext_vector_type(8)))  __bf16 bf16x8;
typedef __attribute__((ext_vector_type(8)))  float  f32x8;
// Matches the builtin's parameter type exactly (per hipcc diagnostic):
typedef int v4i __attribute__((vector_size(16)));

__device__ __forceinline__ bf16x16 cat8(bf16x8 lo, bf16x8 hi) {
  return __builtin_shufflevector(lo, hi, 0,1,2,3,4,5,6,7,8,9,10,11,12,13,14,15);
}

__device__ __forceinline__ void async_wait_all() {
#if HAS_ASYNC_LDS
  __builtin_amdgcn_s_wait_asynccnt(0);
#endif
}

// Copy 16 consecutive elements (convert -> bf16) into LDS. When the source is
// already bf16 and the async-to-LDS path exists, use two B128 async DMAs
// (tracked by ASYNCcnt) instead of VALU loads + ds_store.
template <typename T>
__device__ __forceinline__ void stage16(const T* __restrict__ src, bf16* dst) {
  if constexpr (HAS_ASYNC_LDS != 0 && sizeof(T) == 2) {
#if HAS_ASYNC_LDS
    __builtin_amdgcn_global_load_async_to_lds_b128(
        (__attribute__((address_space(1))) v4i*)(void*)(src),
        (__attribute__((address_space(3))) v4i*)(void*)(dst), 0, 0);
    __builtin_amdgcn_global_load_async_to_lds_b128(
        (__attribute__((address_space(1))) v4i*)(void*)(src + 8),
        (__attribute__((address_space(3))) v4i*)(void*)(dst + 8), 0, 0);
#endif
  } else {
    #pragma unroll
    for (int i = 0; i < 16; ++i) dst[i] = (bf16)(float)src[i];
  }
}

// C[M,N] = alpha * A[M,K] x (BT ? B[N,K]^T : B[K,N]), batched via blockIdx.z.
// bf16 operands, fp32 accumulation (v_wmma_f32_16x16x32_bf16).
// Block tile 128x128 (8 waves, each 32x64 = 8 accumulators), BK=32,
// double-buffered LDS with async global->LDS staging where possible.
// OUT_T: store C transposed per batch of ldcT rows: Ct[b][n][m].
// Dims must divide (128,128,32): holds for every GEMM in this problem.
template <typename TA, typename TB, bool BT, bool OUT_BF16, bool OUT_T>
__global__ __launch_bounds__(256) void gemm_bf16_wmma(
    const TA* __restrict__ A, const TB* __restrict__ B, void* __restrict__ C,
    int N, int K, long sAb, long sBb, long sCb, float alpha, int ldcT)
{
  constexpr int BM = 128, BN = 128, BK = 32;
  __shared__ __align__(16) bf16 shA[2][BM * BK];   // [m][k]
  __shared__ __align__(16) bf16 shB[2][BN * BK];   // [n][k]

  const long bz = blockIdx.z;
  A += bz * sAb;
  B += bz * sBb;
  const int m0 = blockIdx.y * BM;
  const int n0 = blockIdx.x * BN;
  const int tid  = threadIdx.x;
  const int lane = tid & 31;
  const int wave = tid >> 5;
  const int wm = wave & 3, wn = wave >> 2;     // 4 waves on M, 2 on N
  const int half = lane >> 4, r = lane & 15;

  // staging coordinates: 128 rows x 32 k, 16 elems per thread
  const int sr = tid >> 1, sc = (tid & 1) * 16;
  // !BT transpose staging: 32 k x 128 n, 16 elems per thread
  const int tk = tid >> 3, tn = (tid & 7) * 16;

  f32x8 acc[2][4] = {};

  auto stageAB = [&](int k0, int buf) {
    stage16(A + (long)(m0 + sr) * K + (k0 + sc), &shA[buf][sr * BK + sc]);
    if constexpr (BT) {           // B stored [N][K]: direct rows
      stage16(B + (long)(n0 + sr) * K + (k0 + sc), &shB[buf][sr * BK + sc]);
    } else {                      // B stored [K][N]: transpose into [n][k]
      const TB* s = B + (long)(k0 + tk) * N + (n0 + tn);
      #pragma unroll
      for (int i = 0; i < 16; ++i)
        shB[buf][(tn + i) * BK + tk] = (bf16)(float)s[i];
    }
  };

  auto compute = [&](int buf) {
    // A fragment (ISA 7.12.2): lane(r,h) row r, K={8h..8h+7, 16+8h..16+8h+7}
    bf16x16 fa[2], fb[4];
    #pragma unroll
    for (int mi = 0; mi < 2; ++mi) {
      const bf16* p = &shA[buf][(wm * 32 + mi * 16 + r) * BK + half * 8];
      fa[mi] = cat8(*(const bf16x8*)p, *(const bf16x8*)(p + 16));
    }
    // B fragment: lane(r,h) col r, K={8h..8h+7, 16+8h..16+8h+7}
    #pragma unroll
    for (int ni = 0; ni < 4; ++ni) {
      const bf16* p = &shB[buf][(wn * 64 + ni * 16 + r) * BK + half * 8];
      fb[ni] = cat8(*(const bf16x8*)p, *(const bf16x8*)(p + 16));
    }
    #pragma unroll
    for (int mi = 0; mi < 2; ++mi)
      #pragma unroll
      for (int ni = 0; ni < 4; ++ni)
        acc[mi][ni] = __builtin_amdgcn_wmma_f32_16x16x32_bf16(
            false, fa[mi], false, fb[ni], (short)0, acc[mi][ni], false, false);
  };

  stageAB(0, 0);
  async_wait_all();
  __syncthreads();
  int buf = 0;
  for (int k0 = 0; k0 < K; k0 += BK) {
    const int kn = k0 + BK;
    if (kn < K) stageAB(kn, buf ^ 1);   // prefetch next tile (async DMA)
    compute(buf);                       // WMMAs on current tile
    async_wait_all();
    __syncthreads();
    buf ^= 1;
  }

  // C layout: VGPR j -> row (j + 8*half), col r within each 16x16 tile.
  #pragma unroll
  for (int mi = 0; mi < 2; ++mi)
    #pragma unroll
    for (int ni = 0; ni < 4; ++ni) {
      const int row0 = m0 + wm * 32 + mi * 16 + half * 8;
      const int col  = n0 + wn * 64 + ni * 16 + r;
      if constexpr (OUT_T) {
        // per batch of ldcT rows, store Ct[b][col][m]: contiguous in m
        const long b2   = row0 / ldcT;
        const long mloc = row0 - b2 * ldcT;
        const long base = b2 * ((long)N * ldcT) + (long)col * ldcT + mloc;
        #pragma unroll
        for (int j = 0; j < 8; ++j) {
          const float v = acc[mi][ni][j] * alpha;
          if (OUT_BF16) ((bf16*)C)[base + j] = (bf16)v;
          else          ((float*)C)[base + j] = v;
        }
      } else {
        #pragma unroll
        for (int j = 0; j < 8; ++j) {
          const float v = acc[mi][ni][j] * alpha;
          const long idx = bz * sCb + (long)(row0 + j) * N + col;
          if (OUT_BF16) ((bf16*)C)[idx] = (bf16)v;
          else          ((float*)C)[idx] = v;
        }
      }
    }
}

// Row-wise masked softmax: probs_bf16 = softmax(mask>0 ? logits : -9e15).
__global__ __launch_bounds__(256) void softmax_mask_bf16(
    const float* __restrict__ logits, const int* __restrict__ mask,
    bf16* __restrict__ probs, int S)
{
  const long row = blockIdx.x;
  const float* lr = logits + row * (long)S;
  const int*   mr = mask   + row * (long)S;
  bf16*        pr = probs  + row * (long)S;
  const int tid = threadIdx.x;
  __shared__ float red[8];

  float m = -3.0e38f;
  for (int j = tid; j < S; j += 256) {
    const float v = (mr[j] > 0) ? lr[j] : -9.0e15f;
    m = fmaxf(m, v);
  }
  #pragma unroll
  for (int off = 16; off > 0; off >>= 1)
    m = fmaxf(m, __shfl_xor(m, off, 32));        // wave32 reduction
  if ((tid & 31) == 0) red[tid >> 5] = m;
  __syncthreads();
  m = red[0];
  #pragma unroll
  for (int i = 1; i < 8; ++i) m = fmaxf(m, red[i]);
  __syncthreads();

  float s = 0.0f;
  for (int j = tid; j < S; j += 256) {
    const float v = (mr[j] > 0) ? lr[j] : -9.0e15f;
    s += __expf(v - m);
  }
  #pragma unroll
  for (int off = 16; off > 0; off >>= 1)
    s += __shfl_xor(s, off, 32);
  if ((tid & 31) == 0) red[tid >> 5] = s;
  __syncthreads();
  s = red[0] + red[1] + red[2] + red[3] + red[4] + red[5] + red[6] + red[7];
  const float inv = 1.0f / s;

  for (int j = tid; j < S; j += 256) {
    const float v = (mr[j] > 0) ? lr[j] : -9.0e15f;
    pr[j] = (bf16)(__expf(v - m) * inv);
  }
}

extern "C" void kernel_launch(void* const* d_in, const int* in_sizes, int n_in,
                              void* d_out, int out_size, void* d_ws, size_t ws_size,
                              hipStream_t stream)
{
  const int Bb = 8, S = 2048, D = 1024, Ad = 1024, Od = 1024;
  const long MS = (long)Bb * S;   // 16384 flattened rows

  const float* x     = (const float*)d_in[0];
  const int*   amask = (const int*)d_in[1];
  const float* W_K   = (const float*)d_in[2];
  const float* W_Q   = (const float*)d_in[3];
  const float* W_V   = (const float*)d_in[4];
  const float* W_out = (const float*)d_in[5];
  float* out = (float*)d_out;

  // Workspace layout (320 MB total):
  bf16* Kb    = (bf16*)d_ws;            // [16384,1024] bf16
  bf16* Qb    = Kb + MS * Ad;           // [16384,1024] bf16
  bf16* Vt    = Qb + MS * Ad;           // [8][1024][2048] bf16 (transposed V)
  bf16* attnb = Vt + MS * Ad;           // [16384,1024] bf16
  float* logits = (float*)(attnb + MS * Ad);           // [8,2048,2048] f32
  bf16*  probs  = (bf16*)(logits + (long)Bb * S * S);  // [8,2048,2048] bf16

  dim3 blk(256);

  // K/Q projections: [16384,1024] = x @ W -> bf16 row-major
  dim3 gp(Ad / 128, MS / 128, 1);
  gemm_bf16_wmma<float, float, false, true, false><<<gp, blk, 0, stream>>>(
      x, W_K, Kb, Ad, D, 0, 0, 0, 1.0f, 1);
  gemm_bf16_wmma<float, float, false, true, false><<<gp, blk, 0, stream>>>(
      x, W_Q, Qb, Ad, D, 0, 0, 0, 1.0f, 1);
  // V projection: stored transposed per batch -> Vt[b][a][j]
  gemm_bf16_wmma<float, float, false, true, true><<<gp, blk, 0, stream>>>(
      x, W_V, Vt, Ad, D, 0, 0, 0, 1.0f, S);

  // logits[b,s,t] = (1/sqrt(A)) * K[b,s,:] . Q[b,t,:]   (B-transposed, async)
  dim3 gs(S / 128, S / 128, Bb);
  const float inv_sqrt_a = 0.03125f;  // 1/sqrt(1024)
  gemm_bf16_wmma<bf16, bf16, true, false, false><<<gs, blk, 0, stream>>>(
      Kb, Qb, logits, S, Ad, (long)S * Ad, (long)S * Ad, (long)S * S,
      inv_sqrt_a, 1);

  // masked softmax over last dim -> bf16 probs
  softmax_mask_bf16<<<dim3((unsigned)(Bb * S)), blk, 0, stream>>>(
      logits, amask, probs, S);

  // attn[b,s,:] = probs[b,s,:] @ V[b]  (B = Vt, transposed path, async)
  dim3 ga(Ad / 128, S / 128, Bb);
  gemm_bf16_wmma<bf16, bf16, true, true, false><<<ga, blk, 0, stream>>>(
      probs, Vt, attnb, Ad, S, (long)S * S, (long)Ad * S, (long)S * Ad,
      1.0f, 1);

  // out = attn @ W_out  -> fp32 d_out
  dim3 go(Od / 128, MS / 128, 1);
  gemm_bf16_wmma<bf16, float, false, false, false><<<go, blk, 0, stream>>>(
      attnb, W_out, out, Od, Ad, 0, 0, 0, 1.0f, 1);
}
